// GCN_46084999086083
// MI455X (gfx1250) — matrix operand order repaired and checked by reading, not compiled, not run on hardware
//
#include <hip/hip_runtime.h>

typedef __attribute__((ext_vector_type(16))) __bf16 v16bf;
typedef __attribute__((ext_vector_type(8)))  float  v8f;

constexpr int kB = 8;
constexpr int kN = 2048;
constexpr int kC = 256;

// --- aggregation (adj @ support) tiling ---
constexpr int BM = 64;    // adj rows per block
constexpr int BN = 128;   // output cols per block
constexpr int BK = 32;    // K per WMMA step
constexpr int AS = 56;    // LDS row stride (bf16 units) = 112B: 16B-aligned, bank-friendly
constexpr int BS = 56;

// --- support (x @ W) tiling ---
constexpr int XS = 264;   // LDS row stride for 256-wide x tile (528B)

__device__ __forceinline__ __bf16 to_bf16(float f) { return (__bf16)f; }

// Build a v16bf WMMA fragment from two contiguous 16-byte chunks.
__device__ __forceinline__ v16bf frag2(const __bf16* p0, const __bf16* p1) {
  v16bf f;
  ((uint4*)&f)[0] = *(const uint4*)p0;
  ((uint4*)&f)[1] = *(const uint4*)p1;
  return f;
}

// fp32x4 -> bf16x4 packed LDS store (8 bytes)
__device__ __forceinline__ void store_a4(__bf16* dst, float4 v) {
  union { __bf16 hx[4]; uint2 u; } p;
  p.hx[0] = to_bf16(v.x); p.hx[1] = to_bf16(v.y);
  p.hx[2] = to_bf16(v.z); p.hx[3] = to_bf16(v.w);
  *(uint2*)dst = p.u;
}

// gfx1250 async copy: 16B global -> LDS, tracked by ASYNCcnt (ISA §15.18.3 op 98).
// LDS address = low 32 bits of the generic pointer (flat->LDS aperture truncation).
__device__ __forceinline__ void async_b128(__bf16* lds_dst, const __bf16* gsrc) {
  const uint32_t ldsa = (uint32_t)(uintptr_t)lds_dst;
  asm volatile("global_load_async_to_lds_b128 %0, %1, off"
               :: "v"(ldsa), "v"(gsrc) : "memory");
}
__device__ __forceinline__ void wait_async0() {
  asm volatile("s_wait_asynccnt 0" ::: "memory");
}

// ---------------------------------------------------------------------------
// Kernel 1: d_inv_sqrt[b,n] = rsqrt(sum_k adj[b,n,k] + 1e-5). One wave per row.
// ---------------------------------------------------------------------------
__global__ __launch_bounds__(256) void gcn_degree(const float* __restrict__ adj,
                                                  float* __restrict__ dsin) {
  const int row  = blockIdx.x * 8 + (threadIdx.x >> 5);   // 8 waves -> 8 rows
  const int lane = threadIdx.x & 31;
  const float* p = adj + (size_t)row * kN;
  float s = 0.f;
#pragma unroll
  for (int it = 0; it < kN / 128; ++it) {
    const float4 v = *(const float4*)(p + it * 128 + lane * 4);
    s += v.x + v.y + v.z + v.w;
  }
#pragma unroll
  for (int off = 16; off > 0; off >>= 1) s += __shfl_xor(s, off, 32);
  if (lane == 0) dsin[row] = rsqrtf(s + 1e-5f);
}

// ---------------------------------------------------------------------------
// Kernel 2: Wt[o][i] = bf16(W[i][o])  (transposed bf16 copy of the weight)
// ---------------------------------------------------------------------------
__global__ __launch_bounds__(256) void gcn_wt(const float* __restrict__ w,
                                              __bf16* __restrict__ wt) {
  const int idx = blockIdx.x * 256 + threadIdx.x;   // 65536 elements
  const int i = idx >> 8, o = idx & 255;
  wt[(size_t)o * kC + i] = to_bf16(w[(size_t)i * kC + o]);
}

// ---------------------------------------------------------------------------
// Kernel 3: supT[b][o][m] = bf16( d_inv_sqrt[b][m] * (x[b][m][:] . W[:][o]) )
// Block: 16 nodes x 128 outputs (8 waves, one 16x16 tile each), K=256 in 8 steps.
// ---------------------------------------------------------------------------
__global__ __launch_bounds__(256) void gcn_support(const float* __restrict__ x,
                                                   const __bf16* __restrict__ wt,
                                                   const float* __restrict__ dsin,
                                                   __bf16* __restrict__ supT) {
  __shared__ __bf16 Xs[16 * XS];
  const int bz    = blockIdx.z;
  const int node0 = blockIdx.x * 16;
  const int o0    = blockIdx.y * 128;
  const int tid   = threadIdx.x;
  const int lane  = tid & 31;
  const int wid   = tid >> 5;
  const int nloc  = lane & 15;
  const int h     = lane >> 4;

  // Stage 16x256 fp32 x-tile -> bf16 LDS (1024 float4 slots / 256 threads)
  const float* xB = x + ((size_t)bz * kN + node0) * kC;
#pragma unroll
  for (int s = 0; s < 4; ++s) {
    const int slot = tid + s * 256;
    const int r = slot >> 6, c4 = slot & 63;
    store_a4(&Xs[r * XS + c4 * 4], *(const float4*)(xB + (size_t)r * kC + c4 * 4));
  }
  __syncthreads();

  const int o = o0 + wid * 16 + nloc;        // this lane's output column
  v8f acc = {};
#pragma unroll
  for (int k0 = 0; k0 < kC; k0 += BK) {
    const v16bf a = frag2(&Xs[nloc * XS + k0 + h * 8],
                          &Xs[nloc * XS + k0 + 16 + h * 8]);
    const __bf16* wp = wt + (size_t)o * kC + k0 + h * 8;
    const v16bf b = frag2(wp, wp + 16);
    acc = __builtin_amdgcn_wmma_f32_16x16x32_bf16(false, a, false, b,
                                                  (short)0, acc, false, false);
  }

  // Scale rows by d_inv_sqrt[node], store transposed: contiguous b128 per lane.
  union { __bf16 hx[8]; uint4 u; } outp;
#pragma unroll
  for (int r = 0; r < 8; ++r) {
    const int m = node0 + h * 8 + r;
    outp.hx[r] = to_bf16(dsin[(size_t)bz * kN + m] * acc[r]);
  }
  *(uint4*)(supT + ((size_t)bz * kC + o) * kN + node0 + h * 8) = outp.u;
}

// ---------------------------------------------------------------------------
// Kernel 4: out[b][m][n] = dsin[b][m] * (adj[b][m][:] . supT[b][n][:]) + bias[n]
// 64x128 tile per block; 8 waves in 2x4; 2x2 16x16 WMMA tiles per wave; K=2048.
// Double-buffered LDS pipeline: B tile via global_load_async_to_lds_b128,
// A tile via fp32 loads + bf16 convert; s_wait_asynccnt + barrier per stage.
// ---------------------------------------------------------------------------
__global__ __launch_bounds__(256) void gcn_agg(const float* __restrict__ adj,
                                               const __bf16* __restrict__ supT,
                                               const float* __restrict__ dsin,
                                               const float* __restrict__ bias,
                                               float* __restrict__ out) {
  __shared__ __bf16 As[2][BM * AS];   // adj tile, bf16, row-major [m][k]
  __shared__ __bf16 Bs[2][BN * BS];   // supT tile, [n][k] (k contiguous)
  const int bz  = blockIdx.z;
  const int m0  = blockIdx.x * BM;
  const int n0  = blockIdx.y * BN;
  const int tid = threadIdx.x;
  const int lane = tid & 31;
  const int wid  = tid >> 5;
  const int wm   = wid >> 2;       // 0..1 (M direction)
  const int wn   = wid & 3;        // 0..3 (N direction)
  const int nloc = lane & 15;
  const int h    = lane >> 4;

  const float*  adjB = adj  + (size_t)bz * kN * kN;
  const __bf16* supB = supT + (size_t)bz * kC * kN;

  // A-staging slots: 512 float4 (64 rows x 8), 2 per thread
  const int ar  = tid >> 3, ac4 = tid & 7;
  const int ar2 = (tid + 256) >> 3;
  // B-staging slots: 512 x 16B (128 rows x 4), 2 per thread
  const int br  = tid >> 2, bq = tid & 3;
  const int br2 = (tid + 256) >> 2;

  const float*  aP0 = adjB + (size_t)(m0 + ar)  * kN + ac4 * 4;
  const float*  aP1 = adjB + (size_t)(m0 + ar2) * kN + ac4 * 4;
  const __bf16* bP0 = supB + (size_t)(n0 + br)  * kN + bq * 8;
  const __bf16* bP1 = supB + (size_t)(n0 + br2) * kN + bq * 8;

  __bf16* const asd0 = &As[0][ar  * AS + ac4 * 4];
  __bf16* const asd1 = &As[0][ar2 * AS + ac4 * 4];
  __bf16* const bsd0 = &Bs[0][br  * BS + bq * 8];
  __bf16* const bsd1 = &Bs[0][br2 * BS + bq * 8];
  const int abufstride = BM * AS;
  const int bbufstride = BN * BS;

  v8f acc[2][2] = {};

  // ---- prologue: stage K-tile 0 into buffer 0 ----
  {
    async_b128(bsd0, bP0);
    async_b128(bsd1, bP1);
    store_a4(asd0, *(const float4*)aP0);
    store_a4(asd1, *(const float4*)aP1);
  }
  wait_async0();
  __syncthreads();

  for (int k0 = 0; k0 < kN; k0 += BK) {
    const int  buf  = (k0 >> 5) & 1;
    const int  nxt  = buf ^ 1;
    const bool more = (k0 + BK) < kN;

    float4 v0, v1;
    if (more) {
      // issue next stage: async B copy + A global loads (overlap with WMMA below)
      v0 = *(const float4*)(aP0 + k0 + BK);
      v1 = *(const float4*)(aP1 + k0 + BK);
      async_b128(bsd0 + nxt * bbufstride, bP0 + k0 + BK);
      async_b128(bsd1 + nxt * bbufstride, bP1 + k0 + BK);
    }

    // ---- compute on current buffer ----
    const __bf16* A_ = As[buf];
    const __bf16* B_ = Bs[buf];
    v16bf af[2], bfr[2];
#pragma unroll
    for (int tm = 0; tm < 2; ++tm) {
      const int m = wm * 32 + tm * 16 + nloc;
      af[tm] = frag2(&A_[m * AS + h * 8], &A_[m * AS + 16 + h * 8]);
    }
#pragma unroll
    for (int tn = 0; tn < 2; ++tn) {
      const int n = wn * 32 + tn * 16 + nloc;
      bfr[tn] = frag2(&B_[n * BS + h * 8], &B_[n * BS + 16 + h * 8]);
    }
#pragma unroll
    for (int tm = 0; tm < 2; ++tm)
#pragma unroll
      for (int tn = 0; tn < 2; ++tn)
        acc[tm][tn] = __builtin_amdgcn_wmma_f32_16x16x32_bf16(
            false, af[tm], false, bfr[tn], (short)0, acc[tm][tn], false, false);

    if (more) {
      store_a4(asd0 + nxt * abufstride, v0);
      store_a4(asd1 + nxt * abufstride, v1);
    }
    wait_async0();
    __syncthreads();
  }

  // ---- epilogue: row scale + bias; non-temporal fp32 stores (write-once) ----
#pragma unroll
  for (int tm = 0; tm < 2; ++tm) {
    float ds[8];
#pragma unroll
    for (int r = 0; r < 8; ++r)
      ds[r] = dsin[(size_t)bz * kN + m0 + wm * 32 + tm * 16 + h * 8 + r];
#pragma unroll
    for (int tn = 0; tn < 2; ++tn) {
      const int ng = n0 + wn * 32 + tn * 16 + nloc;
      const float bn = bias[ng];
#pragma unroll
      for (int r = 0; r < 8; ++r) {
        const int mg = m0 + wm * 32 + tm * 16 + h * 8 + r;
        __builtin_nontemporal_store(ds[r] * acc[tm][tn][r] + bn,
                                    &out[((size_t)bz * kN + mg) * kC + ng]);
      }
    }
  }
}

// ---------------------------------------------------------------------------
extern "C" void kernel_launch(void* const* d_in, const int* in_sizes, int n_in,
                              void* d_out, int out_size, void* d_ws, size_t ws_size,
                              hipStream_t stream) {
  (void)in_sizes; (void)n_in; (void)out_size; (void)ws_size;
  const float* x      = (const float*)d_in[0];   // [B,N,C]
  const float* adj    = (const float*)d_in[1];   // [B,N,N]
  const float* weight = (const float*)d_in[2];   // [C,C]
  const float* bias   = (const float*)d_in[3];   // [C]
  float* out = (float*)d_out;

  // workspace layout
  float*  dsin = (float*)d_ws;                                        // 64 KB
  __bf16* wt   = (__bf16*)((char*)d_ws + (size_t)kB * kN * 4);        // 128 KB
  __bf16* supT = (__bf16*)((char*)d_ws + (size_t)kB * kN * 4
                                       + (size_t)kC * kC * 2);        // 8 MB

  gcn_degree <<<dim3(kB * kN / 8), 256, 0, stream>>>(adj, dsin);
  gcn_wt     <<<dim3(kC * kC / 256), 256, 0, stream>>>(weight, wt);
  gcn_support<<<dim3(kN / 16, kC / 128, kB), 256, 0, stream>>>(x, wt, dsin, supT);
  gcn_agg    <<<dim3(kN / BM, kC / BN, kB), 256, 0, stream>>>(adj, supT, dsin, bias, out);
}
